// MambaAttention_76098230550852
// MI455X (gfx1250) — matrix operand order repaired and checked by reading, not compiled, not run on hardware
//
#include <hip/hip_runtime.h>
#include <hip/hip_bf16.h>

// ---------------------------------------------------------------------------
// Problem constants (from reference)
// ---------------------------------------------------------------------------
constexpr int kB    = 4;
constexpr int kC    = 192;
constexpr int kH    = 64;
constexpr int kW    = 64;
constexpr int kL    = kH * kW;          // 4096
constexpr int kDIN  = 384;
constexpr int kNST  = 16;               // D_STATE
constexpr int kDTR  = 12;               // DT_RANK
constexpr int kM    = kB * kL;          // 16384 rows for all GEMMs
constexpr float kEPS = 1e-5f;

typedef _Float16 v8h  __attribute__((ext_vector_type(8)));
typedef _Float16 v16h __attribute__((ext_vector_type(16)));
typedef float    v8f  __attribute__((ext_vector_type(8)));

__device__ __forceinline__ int snake_idx(int l) {
  int h = l >> 6, w = l & 63;
  return (h & 1) ? ((h << 6) + (63 - w)) : l;
}

// AS3 (LDS) byte offset from a generic pointer to a __shared__ object:
// addrspacecast(AS3->generic) is {aperture_hi32, lds_offset}, so the low
// 32 bits of the flat address are the DS address.
__device__ __forceinline__ unsigned lds_off_u32(const void* p) {
  return (unsigned)(unsigned long long)p;
}

// CDNA5 async global->LDS 16B transfer (ASYNCcnt-tracked, no VGPR staging)
__device__ __forceinline__ void async_load_b128(unsigned lds_addr, const void* gaddr) {
  asm volatile("global_load_async_to_lds_b128 %0, %1, off"
               :: "v"(lds_addr), "v"((unsigned long long)gaddr)
               : "memory");
}
__device__ __forceinline__ void wait_asynccnt_0() {
  asm volatile("s_wait_asynccnt 0x0" ::: "memory");
}
__device__ __forceinline__ void wait_asynccnt_3() {
  asm volatile("s_wait_asynccnt 0x3" ::: "memory");
}

// ---------------------------------------------------------------------------
// GroupNorm(1, C): per-batch mean / rstd over C*H*W
// ---------------------------------------------------------------------------
__global__ __launch_bounds__(256) void gn_stats_kernel(
    const float* __restrict__ x, float* __restrict__ stats) {
  const int b = blockIdx.x;
  const int tid = threadIdx.x;
  const size_t n = (size_t)kC * kL;            // 786432
  const float* px = x + (size_t)b * n;
  float s = 0.f, s2 = 0.f;
  for (size_t i = tid; i < n; i += 256) {
    float v = px[i];
    s += v; s2 += v * v;
  }
  __shared__ float rs[256], rq[256];
  rs[tid] = s; rq[tid] = s2;
  __syncthreads();
  for (int o = 128; o > 0; o >>= 1) {
    if (tid < o) { rs[tid] += rs[tid + o]; rq[tid] += rq[tid + o]; }
    __syncthreads();
  }
  if (tid == 0) {
    float mu  = rs[0] / (float)n;
    float var = rq[0] / (float)n - mu * mu;
    stats[b * 2 + 0] = mu;
    stats[b * 2 + 1] = rsqrtf(var + kEPS);
  }
}

// normalize + transpose [B,C,HW] -> f16 [B*L, C]
__global__ __launch_bounds__(256) void gn_norm_kernel(
    const float* __restrict__ x, const float* __restrict__ stats,
    const float* __restrict__ gamma, const float* __restrict__ beta,
    _Float16* __restrict__ xs_h) {
  const int total = kB * kC * kL;
  for (int i = blockIdx.x * blockDim.x + threadIdx.x; i < total;
       i += gridDim.x * blockDim.x) {
    int b = i / (kC * kL);
    int rem = i - b * (kC * kL);
    int c = rem / kL;
    int hw = rem & (kL - 1);
    float mu = stats[b * 2 + 0], rstd = stats[b * 2 + 1];
    float v = (x[i] - mu) * rstd * gamma[c] + beta[c];
    xs_h[((size_t)(b * kL + hw)) * kC + c] = (_Float16)v;
  }
}

// f32 [Ns,Ks] -> f16 [Nd,Kd] with zero padding (rows & K)
__global__ __launch_bounds__(256) void cvt_pad_kernel(
    const float* __restrict__ src, _Float16* __restrict__ dst,
    int Ns, int Ks, int Nd, int Kd) {
  int total = Nd * Kd;
  for (int i = blockIdx.x * blockDim.x + threadIdx.x; i < total;
       i += gridDim.x * blockDim.x) {
    int n = i / Kd, k = i - n * Kd;
    float v = (n < Ns && k < Ks) ? src[(size_t)n * Ks + k] : 0.f;
    dst[i] = (_Float16)v;
  }
}

__global__ __launch_bounds__(256) void fill_h_kernel(_Float16* __restrict__ p, int n) {
  for (int i = blockIdx.x * blockDim.x + threadIdx.x; i < n;
       i += gridDim.x * blockDim.x)
    p[i] = (_Float16)0.f;
}

// ---------------------------------------------------------------------------
// Generic WMMA GEMM:  D[M,N] = A[M,Kp] * W[N,Kp]^T   (both f16, K-contiguous)
// Block = 256 threads (8 waves), tile 128(M) x 64(N) x 32(K).
// Each wave owns a 32x32 output (2 A-frags x 2 B-frags -> 4 WMMA / K-step).
// Double-buffered LDS fed by async global->LDS DMA (ASYNCcnt): steady state
// issues the next tile (3 x b128 per thread) then waits ASYNCcnt<=3 (async
// loads complete in order, so the current tile has landed). Last K-step is
// peeled so the hot loop is branch-free.
// ---------------------------------------------------------------------------
#define GTM 128
#define GTN 64
#define GTK 32

__device__ __forceinline__ void epilogue_store(
    int mode, int m, int n, float v, const float* __restrict__ aux0,
    void* __restrict__ out0, void* __restrict__ out1, void* __restrict__ out2) {
  switch (mode) {
    case 0: {  // in_proj: split u (snake scatter) / z (raster), f16
      int b = m >> 12, l = m & (kL - 1);
      if (n < kDIN) {
        int ls = snake_idx(l);
        ((_Float16*)out0)[((size_t)(b * kL + ls)) * kDIN + n] = (_Float16)v;
      } else {
        ((_Float16*)out1)[((size_t)m) * kDIN + (n - kDIN)] = (_Float16)v;
      }
    } break;
    case 1: {  // x_proj: dt (f16, K-padded to 32) | B | C
      if (n < kDTR)            ((_Float16*)out0)[(size_t)m * 32 + n] = (_Float16)v;
      else if (n < kDTR + kNST)((float*)out1)[(size_t)m * kNST + (n - kDTR)] = v;
      else                     ((float*)out2)[(size_t)m * kNST + (n - kDTR - kNST)] = v;
    } break;
    case 2: {  // dt_proj: + bias, softplus -> delta f32
      float t = v + aux0[n];
      float sp = (t > 20.f) ? t : __logf(1.f + __expf(t));
      ((float*)out0)[(size_t)m * kDIN + n] = sp;
    } break;
    case 3: {  // out_proj: f16 row-major [M, 192]
      ((_Float16*)out0)[(size_t)m * kC + n] = (_Float16)v;
    } break;
    case 4: {  // proj_out: scatter f32 to [B, C, H*W]
      int b = m >> 12, hw = m & (kL - 1);
      ((float*)out0)[((size_t)(b * kC + n)) * kL + hw] = v;
    } break;
  }
}

__global__ __launch_bounds__(256) void gemm_wmma_kernel(
    const _Float16* __restrict__ A, const _Float16* __restrict__ Wt,
    int M, int Nlog, int Kp, int mode, const float* __restrict__ aux0,
    void* __restrict__ out0, void* __restrict__ out1, void* __restrict__ out2) {
  __shared__ __align__(16) _Float16 sA[2][GTM * GTK];
  __shared__ __align__(16) _Float16 sW[2][GTN * GTK];

  const int tid  = threadIdx.x;
  const int lane = tid & 31;
  const int wv   = tid >> 5;
  const int wvm  = wv >> 1;          // 0..3  (32-row strip)
  const int wvn  = wv & 1;           // 0..1  (32-col strip)
  const int bm = blockIdx.y * GTM;
  const int bn = blockIdx.x * GTN;

  // per-thread DMA mapping: A tile 128x32 -> 2 x b128/thread, W 64x32 -> 1
  const int arow = tid >> 1;               // 0..127
  const int acol = (tid & 1) * 16;         // 0 or 16 (halves)
  const int wrow = tid >> 2;               // 0..63
  const int wcol = (tid & 3) * 8;          // 0,8,16,24

  const _Float16* gA = A  + (size_t)(bm + arow) * Kp + acol;
  const _Float16* gW = Wt + (size_t)(bn + wrow) * Kp + wcol;
  unsigned dA[2][2], dW[2];
#pragma unroll
  for (int bfi = 0; bfi < 2; ++bfi) {
    dA[bfi][0] = lds_off_u32(&sA[bfi][arow * GTK + acol]);
    dA[bfi][1] = lds_off_u32(&sA[bfi][arow * GTK + acol + 8]);
    dW[bfi]    = lds_off_u32(&sW[bfi][wrow * GTK + wcol]);
  }

  v8f acc00 = {}, acc01 = {}, acc10 = {}, acc11 = {};

  const int half = lane >> 4;        // K-half select
  const int fr   = lane & 15;        // M (A frag) / N (B frag) index
  const int nK   = Kp / GTK;

  auto issue_tile = [&](int kblk, int bfi) {
    const size_t ko = (size_t)kblk * GTK;
    async_load_b128(dA[bfi][0], gA + ko);
    async_load_b128(dA[bfi][1], gA + ko + 8);
    async_load_b128(dW[bfi],    gW + ko);
  };

  auto compute_tile = [&](const _Float16* bA, const _Float16* bW) {
    // A fragments for row strips +0 and +16 of this wave's 32-row band
    const _Float16* pa0 = bA + (wvm * 32 + fr) * GTK;
    const _Float16* pa1 = bA + (wvm * 32 + 16 + fr) * GTK;
    v8h a0l = *(const v8h*)(pa0 + 8 * half);
    v8h a0h = *(const v8h*)(pa0 + 16 + 8 * half);
    v16h af0 = __builtin_shufflevector(a0l, a0h,
        0,1,2,3,4,5,6,7,8,9,10,11,12,13,14,15);
    v8h a1l = *(const v8h*)(pa1 + 8 * half);
    v8h a1h = *(const v8h*)(pa1 + 16 + 8 * half);
    v16h af1 = __builtin_shufflevector(a1l, a1h,
        0,1,2,3,4,5,6,7,8,9,10,11,12,13,14,15);

    // B fragments for col strips +0 and +16 of this wave's 32-col band
    const _Float16* pb0 = bW + (wvn * 32 + fr) * GTK + 8 * half;
    v8h b0l = *(const v8h*)(pb0);
    v8h b0h = *(const v8h*)(pb0 + 16);
    v16h bf0 = __builtin_shufflevector(b0l, b0h,
        0,1,2,3,4,5,6,7,8,9,10,11,12,13,14,15);
    const _Float16* pb1 = bW + (wvn * 32 + 16 + fr) * GTK + 8 * half;
    v8h b1l = *(const v8h*)(pb1);
    v8h b1h = *(const v8h*)(pb1 + 16);
    v16h bf1 = __builtin_shufflevector(b1l, b1h,
        0,1,2,3,4,5,6,7,8,9,10,11,12,13,14,15);

    acc00 = __builtin_amdgcn_wmma_f32_16x16x32_f16(
        false, af0, false, bf0, (short)0, acc00, false, false);
    acc01 = __builtin_amdgcn_wmma_f32_16x16x32_f16(
        false, af0, false, bf1, (short)0, acc01, false, false);
    acc10 = __builtin_amdgcn_wmma_f32_16x16x32_f16(
        false, af1, false, bf0, (short)0, acc10, false, false);
    acc11 = __builtin_amdgcn_wmma_f32_16x16x32_f16(
        false, af1, false, bf1, (short)0, acc11, false, false);
  };

  // prologue: DMA tile 0 into buffer 0
  issue_tile(0, 0);

  // steady state: branch-free issue/wait/compute with double buffering
  for (int it = 0; it < nK - 1; ++it) {
    const int buf = it & 1;
    issue_tile(it + 1, buf ^ 1);
    wait_asynccnt_3();          // oldest 3 (current tile) have landed
    __syncthreads();
    compute_tile(sA[buf], sW[buf]);
    __syncthreads();            // everyone done reading buf before refill
  }
  // peeled last K-step
  {
    const int buf = (nK - 1) & 1;
    wait_asynccnt_0();
    __syncthreads();
    compute_tile(sA[buf], sW[buf]);
  }

  // C/D layout: VGPR r -> M = r + 8*half, N = lane&15
  const int mb0 = bm + wvm * 32;
  const int nb0 = bn + wvn * 32;
#pragma unroll
  for (int r = 0; r < 8; ++r) {
    int m0 = mb0 + r + 8 * half;
    int m1 = m0 + 16;
    int n0 = nb0 + fr;
    int n1 = nb0 + 16 + fr;
    if (m0 < M) {
      if (n0 < Nlog) epilogue_store(mode, m0, n0, acc00[r], aux0, out0, out1, out2);
      if (n1 < Nlog) epilogue_store(mode, m0, n1, acc01[r], aux0, out0, out1, out2);
    }
    if (m1 < M) {
      if (n0 < Nlog) epilogue_store(mode, m1, n0, acc10[r], aux0, out0, out1, out2);
      if (n1 < Nlog) epilogue_store(mode, m1, n1, acc11[r], aux0, out0, out1, out2);
    }
  }
}

// ---------------------------------------------------------------------------
// Causal depthwise conv1d (width 4) along snake sequence + SiLU -> f16
// ---------------------------------------------------------------------------
__global__ __launch_bounds__(256) void conv1d_silu_kernel(
    const _Float16* __restrict__ u_s, const float* __restrict__ w,
    const float* __restrict__ bias, _Float16* __restrict__ u_c) {
  const int total = kM * kDIN;
  for (int i = blockIdx.x * blockDim.x + threadIdx.x; i < total;
       i += gridDim.x * blockDim.x) {
    int m = i / kDIN, d = i - m * kDIN;
    int b = m >> 12, l = m & (kL - 1);
    float acc = bias[d];
#pragma unroll
    for (int t = 0; t < 4; ++t) {
      int j = l - 3 + t;
      if (j >= 0)
        acc += w[d * 4 + t] * (float)u_s[((size_t)(b * kL + j)) * kDIN + d];
    }
    float s = acc / (1.f + __expf(-acc));
    u_c[i] = (_Float16)s;
  }
}

// ---------------------------------------------------------------------------
// Selective scan: 1 block per batch, 1 thread per channel d, 16 states in regs.
// B_t / C_t staged through LDS each step. Writes un-snaked (raster) output.
// ---------------------------------------------------------------------------
__global__ __launch_bounds__(384) void sel_scan_kernel(
    const float* __restrict__ delta, const _Float16* __restrict__ u_c,
    const float* __restrict__ Bmat, const float* __restrict__ Cmat,
    const float* __restrict__ A_log, const float* __restrict__ Dskip,
    float* __restrict__ y_r) {
  const int b = blockIdx.x;
  const int d = threadIdx.x;
  float Areg[kNST], hst[kNST];
#pragma unroll
  for (int n = 0; n < kNST; ++n) {
    Areg[n] = -__expf(A_log[d * kNST + n]);
    hst[n] = 0.f;
  }
  const float Dsk = Dskip[d];
  __shared__ float sB[kNST], sC[kNST];

  for (int l = 0; l < kL; ++l) {
    const int m = (b << 12) + l;
    __syncthreads();
    if (d < kNST)            sB[d] = Bmat[(size_t)m * kNST + d];
    else if (d < 2 * kNST)   sC[d - kNST] = Cmat[(size_t)m * kNST + (d - kNST)];
    __syncthreads();

    float dt = delta[(size_t)m * kDIN + d];
    float ut = (float)u_c[(size_t)m * kDIN + d];
    float du = dt * ut;
    float y = 0.f;
#pragma unroll
    for (int n = 0; n < kNST; ++n) {
      float dA = __expf(dt * Areg[n]);
      hst[n] = dA * hst[n] + du * sB[n];
      y += hst[n] * sC[n];
    }
    int ls = snake_idx(l);   // involution: store straight to raster order
    y_r[((size_t)((b << 12) + ls)) * kDIN + d] = y + ut * Dsk;
  }
}

// gate: y *= silu(z)   (in place, raster order)
__global__ __launch_bounds__(256) void gate_silu_kernel(
    float* __restrict__ y, const _Float16* __restrict__ z) {
  const int total = kM * kDIN;
  for (int i = blockIdx.x * blockDim.x + threadIdx.x; i < total;
       i += gridDim.x * blockDim.x) {
    float zf = (float)z[i];
    y[i] = y[i] * (zf / (1.f + __expf(-zf)));
  }
}

// depthwise 3x3 residual: out_h = f16(y + conv3x3(y))
__global__ __launch_bounds__(256) void le_conv_kernel(
    const float* __restrict__ y, const float* __restrict__ lw,
    _Float16* __restrict__ y2h) {
  const int total = kM * kDIN;
  for (int i = blockIdx.x * blockDim.x + threadIdx.x; i < total;
       i += gridDim.x * blockDim.x) {
    int m = i / kDIN, d = i - m * kDIN;
    int b = m >> 12, hw = m & (kL - 1);
    int h = hw >> 6, w = hw & 63;
    float acc = 0.f;
#pragma unroll
    for (int ky = 0; ky < 3; ++ky) {
      int hh = h + ky - 1;
      if (hh < 0 || hh >= kH) continue;
#pragma unroll
      for (int kx = 0; kx < 3; ++kx) {
        int ww = w + kx - 1;
        if (ww < 0 || ww >= kW) continue;
        acc += lw[d * 9 + ky * 3 + kx] *
               y[((size_t)((b << 12) + (hh << 6) + ww)) * kDIN + d];
      }
    }
    y2h[i] = (_Float16)(y[i] + acc);
  }
}

// ---------------------------------------------------------------------------
// Host orchestration
// ---------------------------------------------------------------------------
extern "C" void kernel_launch(void* const* d_in, const int* in_sizes, int n_in,
                              void* d_out, int out_size, void* d_ws, size_t ws_size,
                              hipStream_t stream) {
  (void)in_sizes; (void)n_in; (void)out_size; (void)ws_size;

  const float* x         = (const float*)d_in[0];
  const float* gn_gamma  = (const float*)d_in[1];
  const float* gn_beta   = (const float*)d_in[2];
  const float* in_proj_w = (const float*)d_in[3];
  const float* conv1d_w  = (const float*)d_in[4];
  const float* conv1d_b  = (const float*)d_in[5];
  const float* x_proj_w  = (const float*)d_in[6];
  const float* dt_proj_w = (const float*)d_in[7];
  const float* dt_proj_b = (const float*)d_in[8];
  const float* A_log     = (const float*)d_in[9];
  const float* Dskip     = (const float*)d_in[10];
  const float* le_w      = (const float*)d_in[11];
  const float* out_proj_w= (const float*)d_in[12];
  const float* proj_out_w= (const float*)d_in[13];
  float* out = (float*)d_out;

  // -------- workspace layout (all region sizes multiple of 256B) ----------
  char* p = (char*)d_ws;
  size_t off = 0;
  auto take = [&](size_t bytes) { void* r = p + off; off += bytes; return r; };

  float*     stats = (float*)    take(256);
  _Float16*  w_in  = (_Float16*) take((size_t)768 * 192 * 2);
  _Float16*  w_xp  = (_Float16*) take((size_t)64  * 384 * 2);   // 44 -> 64 rows, zero pad
  _Float16*  w_dt  = (_Float16*) take((size_t)384 * 32  * 2);   // K 12 -> 32, zero pad
  _Float16*  w_out = (_Float16*) take((size_t)192 * 384 * 2);
  _Float16*  w_po  = (_Float16*) take((size_t)192 * 192 * 2);
  _Float16*  xs_h  = (_Float16*) take((size_t)kM * kC   * 2);   // reused: dt_h | Bmat | Cmat
  _Float16*  u_s_h = (_Float16*) take((size_t)kM * kDIN * 2);   // reused: y2_h
  _Float16*  z_h   = (_Float16*) take((size_t)kM * kDIN * 2);
  _Float16*  u_c_h = (_Float16*) take((size_t)kM * kDIN * 2);   // reused: oseq_h
  float*     delta = (float*)    take((size_t)kM * kDIN * 4);
  float*     y_r   = (float*)    take((size_t)kM * kDIN * 4);

  // aliases (lifetimes are disjoint on the stream timeline)
  _Float16* dt_h   = xs_h;                                   // 1 MiB
  float*    Bmat   = (float*)((char*)xs_h + (size_t)kM * 32 * 2);
  float*    Cmat   = (float*)((char*)Bmat + (size_t)kM * kNST * 4);
  _Float16* y2_h   = u_s_h;
  _Float16* oseq_h = u_c_h;

  // -------- pipeline --------------------------------------------------------
  // 1) GroupNorm
  gn_stats_kernel<<<kB, 256, 0, stream>>>(x, stats);
  gn_norm_kernel<<<2048, 256, 0, stream>>>(x, stats, gn_gamma, gn_beta, xs_h);

  // 2) weight conversions (f32 -> f16, zero padded)
  cvt_pad_kernel<<<256, 256, 0, stream>>>(in_proj_w,  w_in, 768, 192, 768, 192);
  cvt_pad_kernel<<<256, 256, 0, stream>>>(x_proj_w,   w_xp,  44, 384,  64, 384);
  cvt_pad_kernel<<<256, 256, 0, stream>>>(dt_proj_w,  w_dt, 384,  12, 384,  32);
  cvt_pad_kernel<<<256, 256, 0, stream>>>(out_proj_w, w_out,192, 384, 192, 384);
  cvt_pad_kernel<<<256, 256, 0, stream>>>(proj_out_w, w_po, 192, 192, 192, 192);

  // 3) in_proj GEMM (fused snake scatter for u, raster z)
  gemm_wmma_kernel<<<dim3(768 / GTN, kM / GTM), 256, 0, stream>>>(
      xs_h, w_in, kM, 768, 192, 0, nullptr, u_s_h, z_h, nullptr);

  // zero dt pad columns (whole dt buffer; epilogue fills cols 0..11)
  fill_h_kernel<<<512, 256, 0, stream>>>(dt_h, kM * 32);

  // 4) causal depthwise conv1d + SiLU (snake order)
  conv1d_silu_kernel<<<2048, 256, 0, stream>>>(u_s_h, conv1d_w, conv1d_b, u_c_h);

  // 5) x_proj GEMM (N=44 logical; tile pads to 64)
  gemm_wmma_kernel<<<dim3(1, kM / GTM), 256, 0, stream>>>(
      u_c_h, w_xp, kM, 44, 384, 1, nullptr, dt_h, Bmat, Cmat);

  // 6) dt_proj GEMM + bias + softplus -> delta (f32)
  gemm_wmma_kernel<<<dim3(384 / GTN, kM / GTM), 256, 0, stream>>>(
      dt_h, w_dt, kM, 384, 32, 2, dt_proj_b, delta, nullptr, nullptr);

  // 7) selective scan (writes raster-ordered y + Dskip*u)
  sel_scan_kernel<<<kB, kDIN, 0, stream>>>(delta, u_c_h, Bmat, Cmat, A_log, Dskip, y_r);

  // 8) gate with silu(z), then depthwise 3x3 residual -> f16
  gate_silu_kernel<<<2048, 256, 0, stream>>>(y_r, z_h);
  le_conv_kernel<<<2048, 256, 0, stream>>>(y_r, le_w, y2_h);

  // 9) out_proj GEMM -> f16 [M,192]
  gemm_wmma_kernel<<<dim3(192 / GTN, kM / GTM), 256, 0, stream>>>(
      y2_h, w_out, kM, 192, 384, 3, nullptr, oseq_h, nullptr, nullptr);

  // 10) proj_out GEMM -> scatter f32 [B,C,H,W]
  gemm_wmma_kernel<<<dim3(192 / GTN, kM / GTM), 256, 0, stream>>>(
      oseq_h, w_po, kM, 192, 192, 4, nullptr, out, nullptr, nullptr);
}
